// MultiHeadSelfAttention_23562190586362
// MI455X (gfx1250) — compile-verified, matrix operand order checked by reading, not run on hardware
//
#include <hip/hip_runtime.h>
#include <hip/hip_bf16.h>

typedef __attribute__((ext_vector_type(16))) _Float16 v16h;
typedef __attribute__((ext_vector_type(8)))  _Float16 v8h;
typedef __attribute__((ext_vector_type(8)))  float    v8f;

#define B_  8
#define N_  1024
#define C_  1024
#define H_  16
#define D_  64
#define SCALE_ 0.125f   // 1/sqrt(64)

// ---------------------------------------------------------------------------
// WMMA helper: D = A(16x32 f16) * B(32x16 f16) + C(16x16 f32)
// ---------------------------------------------------------------------------
__device__ __forceinline__ v8f wmma16(v16h a, v16h b, v8f c) {
  return __builtin_amdgcn_wmma_f32_16x16x32_f16(
      /*neg_a=*/false, a, /*neg_b=*/false, b,
      /*c_mod=*/(short)0, c, /*reuse_a=*/false, /*reuse_b=*/false);
}

// ---------------------------------------------------------------------------
// Load a 16x32 f16 A-fragment (or B-fragment from a column-major-over-K
// source, i.e. a source where each logical column is a contiguous K-run).
// ISA 16-bit A layout: lanes 0-15 hold row=lane, K = {0..7, 16..23};
//                      lanes 16-31 hold row=lane-16, K = {8..15, 24..31}.
// base points at (row0, k0); ld = element stride between rows/columns.
// ---------------------------------------------------------------------------
__device__ __forceinline__ v16h load_frag(const _Float16* __restrict__ base,
                                          long ld, int lane) {
  const _Float16* p = base + (long)(lane & 15) * ld + ((lane >> 4) << 3);
  v8h lo = *(const v8h*)(p);        // K = kofs .. kofs+7
  v8h hi = *(const v8h*)(p + 16);   // K = kofs+16 .. kofs+23
  return __builtin_shufflevector(lo, hi, 0,1,2,3,4,5,6,7,8,9,10,11,12,13,14,15);
}

// LDS variant (same layout)
__device__ __forceinline__ v16h load_frag_lds(const _Float16* base, int ld, int lane) {
  const _Float16* p = base + (lane & 15) * ld + ((lane >> 4) << 3);
  v8h lo = *(const v8h*)(p);
  v8h hi = *(const v8h*)(p + 16);
  return __builtin_shufflevector(lo, hi, 0,1,2,3,4,5,6,7,8,9,10,11,12,13,14,15);
}

// ---------------------------------------------------------------------------
// f32 -> f16 conversion
// ---------------------------------------------------------------------------
__global__ void cvt_f32_f16(const float* __restrict__ x, _Float16* __restrict__ y, int n) {
  int i = blockIdx.x * blockDim.x + threadIdx.x;
  int stride = gridDim.x * blockDim.x;
  for (; i < n; i += stride) y[i] = (_Float16)x[i];
}

// ---------------------------------------------------------------------------
// QKV GEMM: qkv[m][o] = sum_c X[m][c] * Wqkv[o][c]
//   X: [B*N, C] f16 row-major, Wqkv: [3C, C] f16 row-major (rows = B columns).
// Block = 256 threads (8 waves in 2Mx4N), block tile 64 x 256,
// wave tile 32(M) x 64(N): per k-step 6 b128-load-pairs feed 8 WMMAs.
// Epilogue scatters into Q,K: [B,H,N,D], V transposed: [B,H,D,N].
// ---------------------------------------------------------------------------
__global__ void __launch_bounds__(256)
gemm_qkv(const _Float16* __restrict__ X, const _Float16* __restrict__ W,
         _Float16* __restrict__ Q, _Float16* __restrict__ K,
         _Float16* __restrict__ Vt) {
  const int lane = threadIdx.x & 31;
  const int wave = threadIdx.x >> 5;
  const int wm = wave >> 2;        // 0..1
  const int wn = wave & 3;         // 0..3
  const long m0 = (long)blockIdx.x * 64 + wm * 32;
  const long n0 = (long)blockIdx.y * 256 + wn * 64;

  v8f acc[2][4] = {};
#pragma unroll 2
  for (int k0 = 0; k0 < C_; k0 += 32) {
    v16h a0 = load_frag(X + (m0     ) * C_ + k0, C_, lane);
    v16h a1 = load_frag(X + (m0 + 16) * C_ + k0, C_, lane);
    v16h b0 = load_frag(W + (n0     ) * C_ + k0, C_, lane);
    v16h b1 = load_frag(W + (n0 + 16) * C_ + k0, C_, lane);
    v16h b2 = load_frag(W + (n0 + 32) * C_ + k0, C_, lane);
    v16h b3 = load_frag(W + (n0 + 48) * C_ + k0, C_, lane);
    acc[0][0] = wmma16(a0, b0, acc[0][0]);
    acc[0][1] = wmma16(a0, b1, acc[0][1]);
    acc[0][2] = wmma16(a0, b2, acc[0][2]);
    acc[0][3] = wmma16(a0, b3, acc[0][3]);
    acc[1][0] = wmma16(a1, b0, acc[1][0]);
    acc[1][1] = wmma16(a1, b1, acc[1][1]);
    acc[1][2] = wmma16(a1, b2, acc[1][2]);
    acc[1][3] = wmma16(a1, b3, acc[1][3]);
  }

#pragma unroll
  for (int i = 0; i < 2; ++i)
#pragma unroll
    for (int j = 0; j < 4; ++j)
#pragma unroll
      for (int r = 0; r < 8; ++r) {
        long m = m0 + i * 16 + r + ((lane >> 4) << 3);
        long o = n0 + j * 16 + (lane & 15);
        int  b = (int)(m >> 10);       // N_ = 1024
        int  n = (int)(m & 1023);
        int  s = (int)(o >> 10);       // C_ = 1024
        int  rem = (int)(o & 1023);
        int  h = rem >> 6;             // D_ = 64
        int  d = rem & 63;
        long bh = (long)b * H_ + h;
        _Float16 val = (_Float16)acc[i][j][r];
        if (s == 0)      Q [(bh * N_ + n) * D_ + d] = val;
        else if (s == 1) K [(bh * N_ + n) * D_ + d] = val;
        else             Vt[(bh * D_ + d) * N_ + n] = val;
      }
}

// ---------------------------------------------------------------------------
// Flash attention: one wave handles 16 query rows, streams keys 32 at a time.
// Block = 128 threads (4 independent waves). LDS stages P (C-layout -> A-layout).
// Output written as f16 into [B, N, C] (C = h*64 + d) for the final GEMM.
// ---------------------------------------------------------------------------
__global__ void __launch_bounds__(128)
attn_fwd(const _Float16* __restrict__ Q, const _Float16* __restrict__ K,
         const _Float16* __restrict__ Vt, _Float16* __restrict__ O) {
  __shared__ _Float16 Pl[4][16 * 32];
  const int lane = threadIdx.x & 31;
  const int wave = threadIdx.x >> 5;
  const int gq = blockIdx.x * 4 + wave;   // [0, B*H*(N/16))
  const int qb = gq & 63;                 // q-block within (b,h)
  const int bh = gq >> 6;                 // 0..B*H-1

  const _Float16* Qp = Q  + ((long)bh * N_ + qb * 16) * D_;
  const _Float16* Kp = K  + (long)bh * N_ * D_;
  const _Float16* Vp = Vt + (long)bh * D_ * N_;

  // Q fragments: 16 rows x 64 d, split into two K=32 fragments.
  v16h qa0 = load_frag(Qp + 0,  D_, lane);
  v16h qa1 = load_frag(Qp + 32, D_, lane);

  v8f  o_acc[4] = {};          // 16 x 64 accumulator (4 d-tiles)
  float mrun[8], lrun[8];
#pragma unroll
  for (int r = 0; r < 8; ++r) { mrun[r] = -1e30f; lrun[r] = 0.f; }

#pragma unroll 1
  for (int kb = 0; kb < N_; kb += 32) {
    // ---- S = Q . K^T for 32 keys: two 16x16 tiles -----------------------
    v8f s0 = {}, s1 = {};
    {
      v16h kb0 = load_frag(Kp + (long)(kb     ) * D_ + 0,  D_, lane);
      v16h kb1 = load_frag(Kp + (long)(kb     ) * D_ + 32, D_, lane);
      s0 = wmma16(qa0, kb0, s0);
      s0 = wmma16(qa1, kb1, s0);
      v16h kc0 = load_frag(Kp + (long)(kb + 16) * D_ + 0,  D_, lane);
      v16h kc1 = load_frag(Kp + (long)(kb + 16) * D_ + 32, D_, lane);
      s1 = wmma16(qa0, kc0, s1);
      s1 = wmma16(qa1, kc1, s1);
    }

    // ---- online softmax over this 32-key block --------------------------
    float alpha[8];
#pragma unroll
    for (int r = 0; r < 8; ++r) {
      float v0 = s0[r] * SCALE_;
      float v1 = s1[r] * SCALE_;
      float mx = fmaxf(v0, v1);
      mx = fmaxf(mx, __shfl_xor(mx, 1));
      mx = fmaxf(mx, __shfl_xor(mx, 2));
      mx = fmaxf(mx, __shfl_xor(mx, 4));
      mx = fmaxf(mx, __shfl_xor(mx, 8));     // row max within 16-lane group
      float mnew = fmaxf(mrun[r], mx);
      float p0 = __expf(v0 - mnew);
      float p1 = __expf(v1 - mnew);
      float rs = p0 + p1;
      rs += __shfl_xor(rs, 1);
      rs += __shfl_xor(rs, 2);
      rs += __shfl_xor(rs, 4);
      rs += __shfl_xor(rs, 8);               // row sum within 16-lane group
      alpha[r] = __expf(mrun[r] - mnew);
      lrun[r]  = lrun[r] * alpha[r] + rs;
      mrun[r]  = mnew;
      s0[r] = p0;
      s1[r] = p1;
    }
#pragma unroll
    for (int t = 0; t < 4; ++t)
#pragma unroll
      for (int r = 0; r < 8; ++r) o_acc[t][r] *= alpha[r];

    // ---- stage P (16x32 f16) through LDS: C-layout -> A-layout ----------
    __syncthreads();
#pragma unroll
    for (int r = 0; r < 8; ++r) {
      int m = r + ((lane >> 4) << 3);
      Pl[wave][m * 32 +      (lane & 15)] = (_Float16)s0[r];
      Pl[wave][m * 32 + 16 + (lane & 15)] = (_Float16)s1[r];
    }
    __syncthreads();
    v16h pa = load_frag_lds(&Pl[wave][0], 32, lane);

    // ---- O += P . V  (V stored transposed: [D, N]) ----------------------
#pragma unroll
    for (int t = 0; t < 4; ++t) {
      v16h vb = load_frag(Vp + (long)(t * 16) * N_ + kb, N_, lane);
      o_acc[t] = wmma16(pa, vb, o_acc[t]);
    }
  }

  // ---- epilogue: normalize and write f16 into [B, N, C] -----------------
  const int b = bh >> 4;
  const int h = bh & 15;
#pragma unroll
  for (int t = 0; t < 4; ++t)
#pragma unroll
    for (int r = 0; r < 8; ++r) {
      int m   = r + ((lane >> 4) << 3);
      long row = (long)b * N_ + qb * 16 + m;
      int  col = h * D_ + t * 16 + (lane & 15);
      O[row * C_ + col] = (_Float16)(o_acc[t][r] / lrun[r]);
    }
}

// ---------------------------------------------------------------------------
// Output projection: out[m][o] = sum_c A[m][c] * Wp[o][c] + bias[o]  (f32 out)
// Same 32x64 wave tile / 64x256 block tile as gemm_qkv.
// ---------------------------------------------------------------------------
__global__ void __launch_bounds__(256)
gemm_proj(const _Float16* __restrict__ A, const _Float16* __restrict__ W,
          const float* __restrict__ bias, float* __restrict__ out) {
  const int lane = threadIdx.x & 31;
  const int wave = threadIdx.x >> 5;
  const int wm = wave >> 2;
  const int wn = wave & 3;
  const long m0 = (long)blockIdx.x * 64 + wm * 32;
  const long n0 = (long)blockIdx.y * 256 + wn * 64;

  v8f acc[2][4] = {};
#pragma unroll 2
  for (int k0 = 0; k0 < C_; k0 += 32) {
    v16h a0 = load_frag(A + (m0     ) * C_ + k0, C_, lane);
    v16h a1 = load_frag(A + (m0 + 16) * C_ + k0, C_, lane);
    v16h b0 = load_frag(W + (n0     ) * C_ + k0, C_, lane);
    v16h b1 = load_frag(W + (n0 + 16) * C_ + k0, C_, lane);
    v16h b2 = load_frag(W + (n0 + 32) * C_ + k0, C_, lane);
    v16h b3 = load_frag(W + (n0 + 48) * C_ + k0, C_, lane);
    acc[0][0] = wmma16(a0, b0, acc[0][0]);
    acc[0][1] = wmma16(a0, b1, acc[0][1]);
    acc[0][2] = wmma16(a0, b2, acc[0][2]);
    acc[0][3] = wmma16(a0, b3, acc[0][3]);
    acc[1][0] = wmma16(a1, b0, acc[1][0]);
    acc[1][1] = wmma16(a1, b1, acc[1][1]);
    acc[1][2] = wmma16(a1, b2, acc[1][2]);
    acc[1][3] = wmma16(a1, b3, acc[1][3]);
  }

#pragma unroll
  for (int i = 0; i < 2; ++i)
#pragma unroll
    for (int j = 0; j < 4; ++j)
#pragma unroll
      for (int r = 0; r < 8; ++r) {
        long m = m0 + i * 16 + r + ((lane >> 4) << 3);
        long o = n0 + j * 16 + (lane & 15);
        out[m * C_ + o] = acc[i][j][r] + bias[o];
      }
}

// ---------------------------------------------------------------------------
// Launch
// ---------------------------------------------------------------------------
extern "C" void kernel_launch(void* const* d_in, const int* in_sizes, int n_in,
                              void* d_out, int out_size, void* d_ws, size_t ws_size,
                              hipStream_t stream) {
  const float* x      = (const float*)d_in[0];  // [B,N,C]
  const float* w_qkv  = (const float*)d_in[1];  // [3C,C]
  const float* w_proj = (const float*)d_in[2];  // [C,C]
  const float* b_proj = (const float*)d_in[3];  // [C]
  float* out = (float*)d_out;                   // [B,N,C]

  const long nX    = (long)B_ * N_ * C_;   // 8M
  const long nWqkv = (long)3 * C_ * C_;    // 3M
  const long nWp   = (long)C_ * C_;        // 1M
  const long nBHND = (long)B_ * H_ * N_ * D_; // 8M

  _Float16* xh  = (_Float16*)d_ws;
  _Float16* wqh = xh  + nX;
  _Float16* wph = wqh + nWqkv;
  _Float16* qw  = wph + nWp;
  _Float16* kw  = qw  + nBHND;
  _Float16* vtw = kw  + nBHND;
  _Float16* aw  = vtw + nBHND;   // attention output [B,N,C] f16

  // 1) f32 -> f16 conversions
  cvt_f32_f16<<<(int)((nX    + 255) / 256), 256, 0, stream>>>(x,      xh,  (int)nX);
  cvt_f32_f16<<<(int)((nWqkv + 255) / 256), 256, 0, stream>>>(w_qkv,  wqh, (int)nWqkv);
  cvt_f32_f16<<<(int)((nWp   + 255) / 256), 256, 0, stream>>>(w_proj, wph, (int)nWp);

  // 2) QKV projection: M = B*N = 8192, N = 3C = 3072 (block tile 64x256)
  gemm_qkv<<<dim3(8192 / 64, 3072 / 256), 256, 0, stream>>>(xh, wqh, qw, kw, vtw);

  // 3) Flash attention: B*H*(N/16) q-blocks, 4 waves per block
  attn_fwd<<<(B_ * H_ * (N_ / 16)) / 4, 128, 0, stream>>>(qw, kw, vtw, aw);

  // 4) Output projection: M = 8192, N = C = 1024 (block tile 64x256)
  gemm_proj<<<dim3(8192 / 64, 1024 / 256), 256, 0, stream>>>(aw, wph, b_proj, out);
}
